// GraphSAGE_12481174963004
// MI455X (gfx1250) — matrix operand order repaired and checked by reading, not compiled, not run on hardware
//
#include <hip/hip_runtime.h>
#include <hip/hip_bf16.h>

#define D_FEAT 128

typedef __attribute__((ext_vector_type(16))) __bf16 v16bf;
typedef __attribute__((ext_vector_type(8)))  __bf16 v8bf;
typedef __attribute__((ext_vector_type(8)))  float  v8f;

// ---------- helpers ----------

__device__ __forceinline__ void ld8(const float* __restrict__ p, float* v) {
    float4 a = reinterpret_cast<const float4*>(p)[0];
    float4 b = reinterpret_cast<const float4*>(p)[1];
    v[0]=a.x; v[1]=a.y; v[2]=a.z; v[3]=a.w;
    v[4]=b.x; v[5]=b.y; v[6]=b.z; v[7]=b.w;
}

// fp32 -> bf16 hi (truncate) + bf16 lo (RNE of residual), as raw bits
__device__ __forceinline__ void split_bits(float f, unsigned short& hb, unsigned short& lb) {
    unsigned u = __float_as_uint(f);
    hb = (unsigned short)(u >> 16);
    float hf = __uint_as_float(u & 0xFFFF0000u);
    __bf16 l = (__bf16)(f - hf);
    lb = __builtin_bit_cast(unsigned short, l);
}

__device__ __forceinline__ unsigned pack2(unsigned short a, unsigned short b) {
    return (unsigned)a | ((unsigned)b << 16);
}

// split 8 scaled floats into two uint4 (8 packed bf16 each)
__device__ __forceinline__ void split8(const float* f, float scale, uint4& hiq, uint4& loq) {
    unsigned short hb[8], lb[8];
#pragma unroll
    for (int j = 0; j < 8; ++j) split_bits(f[j] * scale, hb[j], lb[j]);
    hiq = make_uint4(pack2(hb[0],hb[1]), pack2(hb[2],hb[3]), pack2(hb[4],hb[5]), pack2(hb[6],hb[7]));
    loq = make_uint4(pack2(lb[0],lb[1]), pack2(lb[2],lb[3]), pack2(lb[4],lb[5]), pack2(lb[6],lb[7]));
}

__device__ __forceinline__ v16bf cat8(v8bf a, v8bf b) {
    return __builtin_shufflevector(a, b, 0,1,2,3,4,5,6,7,8,9,10,11,12,13,14,15);
}

// ---------- kernels ----------

__global__ void k_degree(const int* __restrict__ dst, float* __restrict__ deg, int E) {
    int t = blockIdx.x * blockDim.x + threadIdx.x;
    if (t < E) atomicAdd(&deg[dst[t]], 1.0f);
}

// One wave per edge: lane c moves floats [4c, 4c+4) of the 128-wide feature row.
__global__ void k_scatter(const float* __restrict__ x, const int* __restrict__ src,
                          const int* __restrict__ dst, float* __restrict__ agg, int E) {
    long t = (long)blockIdx.x * blockDim.x + threadIdx.x;
    long e = t >> 5;
    int  c = (int)(t & 31);
    if (e >= E) return;
    int s = src[e], d = dst[e];
    float4 v = *reinterpret_cast<const float4*>(x + (size_t)s * D_FEAT + c * 4);
    float* o = agg + (size_t)d * D_FEAT + c * 4;
    atomicAdd(o + 0, v.x);
    atomicAdd(o + 1, v.y);
    atomicAdd(o + 2, v.z);
    atomicAdd(o + 3, v.w);
}

// one-shot weight split: f32 [n] -> bf16 hi plane + bf16 lo plane
__global__ void k_split_w(const float* __restrict__ w, __bf16* __restrict__ hi,
                          __bf16* __restrict__ lo, int n) {
    int t = blockIdx.x * blockDim.x + threadIdx.x;
    int i = t * 2;
    if (i >= n) return;
    unsigned short h0, l0, h1, l1;
    split_bits(w[i],     h0, l0);
    split_bits(w[i + 1], h1, l1);
    reinterpret_cast<unsigned*>(hi)[t] = pack2(h0, h1);
    reinterpret_cast<unsigned*>(lo)[t] = pack2(l0, l1);
}

// Fused SAGE layer: out = epilogue( (agg/deg) @ wn^T + x @ ws^T + bs )
// MODE 0: LayerNorm + ReLU.  MODE 1: row L2-normalize.
// Block = NOUT/16 waves covering a 16-row stripe; wave w owns columns [16w,16w+16).
// A-side staged once per block in LDS as pre-split bf16 hi/lo; B-side pre-split in global.
template<int NOUT, int MODE>
__global__ __launch_bounds__(NOUT * 2)
void k_sage_gemm(const float* __restrict__ aggsum, const float* __restrict__ deg,
                 const float* __restrict__ xself,
                 const __bf16* __restrict__ wnh, const __bf16* __restrict__ wnl,
                 const __bf16* __restrict__ wsh, const __bf16* __restrict__ wsl,
                 const float* __restrict__ bs, const float* __restrict__ gamma,
                 const float* __restrict__ beta, float* __restrict__ out, int nrows) {
    constexpr int LDP = D_FEAT + 8;   // padded LDS row (bf16): 272B -> conflict-free frag reads
    __shared__ __bf16 sAh[16][LDP], sAl[16][LDP], sXh[16][LDP], sXl[16][LDP];
    __shared__ float  smem[16][NOUT + 4];

    const int tid  = threadIdx.x;
    const int wave = tid >> 5;
    const int lane = tid & 31;
    const int half = lane >> 4;
    const int l15  = lane & 15;
    const int m0   = blockIdx.x * 16;
    const int n0   = wave * 16;

    // ---- cooperative staging: load 16x128 agg & x rows once, split to bf16 hi/lo ----
    constexpr int NTHR   = NOUT * 2;
    constexpr int CHUNKS = (16 * D_FEAT) / (NTHR * 8);
#pragma unroll
    for (int ch = 0; ch < CHUNKS; ++ch) {
        const int idx = (tid + ch * NTHR) * 8;
        const int row = idx >> 7;
        const int col = idx & (D_FEAT - 1);
        int grow = m0 + row;
        if (grow >= nrows) grow = nrows - 1;
        const float invd = 1.0f / fmaxf(deg[grow], 1.0f);

        float fa[8], fx[8];
        ld8(aggsum + (size_t)grow * D_FEAT + col, fa);
        ld8(xself  + (size_t)grow * D_FEAT + col, fx);

        uint4 h, l;
        split8(fa, invd, h, l);
        *reinterpret_cast<uint4*>(&sAh[row][col]) = h;
        *reinterpret_cast<uint4*>(&sAl[row][col]) = l;
        split8(fx, 1.0f, h, l);
        *reinterpret_cast<uint4*>(&sXh[row][col]) = h;
        *reinterpret_cast<uint4*>(&sXl[row][col]) = l;
    }
    __syncthreads();

    // ---- WMMA main loop: bf16x3 split product, fragments straight from LDS/global ----
    const size_t wrow = (size_t)(n0 + l15) * D_FEAT;
    v8f acc = {};
#pragma unroll
    for (int kb = 0; kb < D_FEAT; kb += 32) {
        const int b1 = kb + half * 16;       // A frag elems 0..7  -> K = b1 + j
        const int b2 = kb + half * 16 + 8;   // A frag elems 8..15 -> K = b2 + (j-8)
        const int bb = kb + half * 16;       // B frag elems 0..15 -> K = bb + j

        v16bf ah = cat8(*(const v8bf*)&sAh[l15][b1], *(const v8bf*)&sAh[l15][b2]);
        v16bf al = cat8(*(const v8bf*)&sAl[l15][b1], *(const v8bf*)&sAl[l15][b2]);
        v16bf xh = cat8(*(const v8bf*)&sXh[l15][b1], *(const v8bf*)&sXh[l15][b2]);
        v16bf xl = cat8(*(const v8bf*)&sXl[l15][b1], *(const v8bf*)&sXl[l15][b2]);

        v16bf nh = *(const v16bf*)(wnh + wrow + bb);
        v16bf nl = *(const v16bf*)(wnl + wrow + bb);
        v16bf sh = *(const v16bf*)(wsh + wrow + bb);
        v16bf sl = *(const v16bf*)(wsl + wrow + bb);

        // neighbor term: (agg/deg) @ wn^T   (hi*hi + hi*lo + lo*hi)
        acc = __builtin_amdgcn_wmma_f32_16x16x32_bf16(false, ah, false, nh, (short)0, acc, false, false);
        acc = __builtin_amdgcn_wmma_f32_16x16x32_bf16(false, ah, false, nl, (short)0, acc, false, false);
        acc = __builtin_amdgcn_wmma_f32_16x16x32_bf16(false, al, false, nh, (short)0, acc, false, false);
        // self term: x @ ws^T
        acc = __builtin_amdgcn_wmma_f32_16x16x32_bf16(false, xh, false, sh, (short)0, acc, false, false);
        acc = __builtin_amdgcn_wmma_f32_16x16x32_bf16(false, xh, false, sl, (short)0, acc, false, false);
        acc = __builtin_amdgcn_wmma_f32_16x16x32_bf16(false, xl, false, sh, (short)0, acc, false, false);
    }

    // ---- stage tile (+bias) in LDS. C layout: VGPR r -> M = r + 8*half, N = l15 ----
    const float bias = bs[n0 + l15];
#pragma unroll
    for (int r = 0; r < 8; ++r) {
        smem[r + half * 8][n0 + l15] = acc[r] + bias;
    }
    __syncthreads();

    // ---- row-wise epilogue: TPR contiguous threads per row, 8 columns each ----
    constexpr int TPR = NOUT / 8;
    constexpr int CPT = 8;
    const int row  = tid / TPR;
    const int c0   = (tid % TPR) * CPT;
    const int grow = m0 + row;

    float vals[CPT];
    float sum = 0.f, sq = 0.f;
#pragma unroll
    for (int j = 0; j < CPT; ++j) {
        float v = smem[row][c0 + j];
        vals[j] = v;
        sum += v;
        sq  += v * v;
    }
#pragma unroll
    for (int off = TPR >> 1; off; off >>= 1) {
        sum += __shfl_xor(sum, off, TPR);
        sq  += __shfl_xor(sq,  off, TPR);
    }

    if (grow < nrows) {
        float* orow = out + (size_t)grow * NOUT;
        if constexpr (MODE == 0) {
            const float mean = sum * (1.0f / NOUT);
            const float var  = sq * (1.0f / NOUT) - mean * mean;
            const float rstd = rsqrtf(var + 1e-5f);
#pragma unroll
            for (int j = 0; j < CPT; ++j) {
                float v = (vals[j] - mean) * rstd * gamma[c0 + j] + beta[c0 + j];
                orow[c0 + j] = fmaxf(v, 0.0f);
            }
        } else {
            const float scale = 1.0f / fmaxf(sqrtf(sq), 1e-12f);
#pragma unroll
            for (int j = 0; j < CPT; ++j) {
                orow[c0 + j] = vals[j] * scale;
            }
        }
    }
}

// ---------- host ----------

extern "C" void kernel_launch(void* const* d_in, const int* in_sizes, int n_in,
                              void* d_out, int out_size, void* d_ws, size_t ws_size,
                              hipStream_t stream) {
    (void)n_in; (void)out_size; (void)ws_size;

    const float* x   = (const float*)d_in[0];
    const int*   ei  = (const int*)  d_in[1];
    const float* wn0 = (const float*)d_in[2];
    const float* ws0 = (const float*)d_in[3];
    const float* bs0 = (const float*)d_in[4];
    const float* g0  = (const float*)d_in[5];
    const float* b0  = (const float*)d_in[6];
    const float* wn1 = (const float*)d_in[7];
    const float* ws1 = (const float*)d_in[8];
    const float* bs1 = (const float*)d_in[9];
    const float* g1  = (const float*)d_in[10];
    const float* b1  = (const float*)d_in[11];
    const float* wn2 = (const float*)d_in[12];
    const float* ws2 = (const float*)d_in[13];
    const float* bs2 = (const float*)d_in[14];

    const int N = in_sizes[0] / D_FEAT;
    const int E = in_sizes[1] / 2;
    const int* src = ei;
    const int* dst = ei + E;
    float* out = (float*)d_out;

    const size_t featBytes = (size_t)N * D_FEAT * sizeof(float);
    const size_t wElems    = (size_t)D_FEAT * D_FEAT;          // max weight size
    char* ws = (char*)d_ws;
    float*  agg = (float*)ws;   ws += featBytes;
    float*  h1  = (float*)ws;   ws += featBytes;
    float*  h2  = (float*)ws;   ws += featBytes;
    float*  deg = (float*)ws;   ws += (size_t)N * sizeof(float);
    __bf16* wnh = (__bf16*)ws;  ws += wElems * 2;
    __bf16* wnl = (__bf16*)ws;  ws += wElems * 2;
    __bf16* wsh = (__bf16*)ws;  ws += wElems * 2;
    __bf16* wsl = (__bf16*)ws;

    const int gemmGrid  = (N + 15) / 16;
    const long scatterT = (long)E * 32;
    const int scatterG  = (int)((scatterT + 255) / 256);
    const int wsplitG   = (int)((wElems / 2 + 255) / 256);

    // degree (shared by all layers)
    hipMemsetAsync(deg, 0, (size_t)N * sizeof(float), stream);
    k_degree<<<(E + 255) / 256, 256, 0, stream>>>(dst, deg, E);

    // layer 0: x -> h1
    k_split_w<<<wsplitG, 256, 0, stream>>>(wn0, wnh, wnl, (int)wElems);
    k_split_w<<<wsplitG, 256, 0, stream>>>(ws0, wsh, wsl, (int)wElems);
    hipMemsetAsync(agg, 0, featBytes, stream);
    k_scatter<<<scatterG, 256, 0, stream>>>(x, src, dst, agg, E);
    k_sage_gemm<128, 0><<<gemmGrid, 256, 0, stream>>>(agg, deg, x, wnh, wnl, wsh, wsl,
                                                      bs0, g0, b0, h1, N);

    // layer 1: h1 -> h2
    k_split_w<<<wsplitG, 256, 0, stream>>>(wn1, wnh, wnl, (int)wElems);
    k_split_w<<<wsplitG, 256, 0, stream>>>(ws1, wsh, wsl, (int)wElems);
    hipMemsetAsync(agg, 0, featBytes, stream);
    k_scatter<<<scatterG, 256, 0, stream>>>(h1, src, dst, agg, E);
    k_sage_gemm<128, 0><<<gemmGrid, 256, 0, stream>>>(agg, deg, h1, wnh, wnl, wsh, wsl,
                                                      bs1, g1, b1, h2, N);

    // layer 2: h2 -> out (L2 normalize), 64 output cols
    const size_t wElems2 = (size_t)64 * D_FEAT;
    const int wsplitG2   = (int)((wElems2 / 2 + 255) / 256);
    k_split_w<<<wsplitG2, 256, 0, stream>>>(wn2, wnh, wnl, (int)wElems2);
    k_split_w<<<wsplitG2, 256, 0, stream>>>(ws2, wsh, wsl, (int)wElems2);
    hipMemsetAsync(agg, 0, featBytes, stream);
    k_scatter<<<scatterG, 256, 0, stream>>>(h2, src, dst, agg, E);
    k_sage_gemm<64, 1><<<gemmGrid, 128, 0, stream>>>(agg, deg, h2, wnh, wnl, wsh, wsl,
                                                     bs2, nullptr, nullptr, out, N);
}
